// AttentionLayer_57629871177829
// MI455X (gfx1250) — matrix operand order repaired and checked by reading, not compiled, not run on hardware
//
#include <hip/hip_runtime.h>
#include <stdint.h>

typedef __attribute__((ext_vector_type(16))) _Float16 v16h;
typedef __attribute__((ext_vector_type(8)))  float    v8f;
typedef __attribute__((ext_vector_type(4)))  int      v4i;

#define CFENCE() asm volatile("" ::: "memory")

#define EDIM  64
#define H1DIM 80
#define H2DIM 40
#define NWAVE 4
#define LPAD  208

// packed-weight region offsets (halves)
#define OFF_Q   0        // (W1a + W1c): 2 slabs x 5 ntiles
#define OFF_K   5120     // (W1b - W1c): 2 x 5
#define OFF_QK  10240    // W1d        : 2 x 5
#define OFF_W2  15360    // W2 padded  : 3 x 3
#define TOTAL_PACK (15360 + 4608)

// ---- CDNA5 async global->LDS path (gated; falls back to sync float4 copy) ----
#if __has_builtin(__builtin_amdgcn_global_load_async_to_lds_b128)
#define HAVE_ASYNC 1
typedef __attribute__((address_space(1))) v4i gv4i;   // global-AS int4
typedef __attribute__((address_space(3))) v4i lv4i;   // LDS-AS int4
__device__ __forceinline__ void async_cp16(const void* g, void* l) {
  __builtin_amdgcn_global_load_async_to_lds_b128((gv4i*)g, (lv4i*)l, 0, 0);
}
#else
#define HAVE_ASYNC 0
#endif

#if HAVE_ASYNC
#if __has_builtin(__builtin_amdgcn_s_wait_asynccnt)
#define TILE_SYNC(N) do { __builtin_amdgcn_s_wait_asynccnt(N); CFENCE(); } while (0)
#else
#define TILE_SYNC(N) do { asm volatile("s_wait_asynccnt %0" :: "i"(N) : "memory"); } while (0)
#endif
#else
#define TILE_SYNC(N) CFENCE()
#endif

// ---------------------------------------------------------------------------
// Pack folded W1 blocks and W2 into f16 WMMA B-fragment layout for
// v_wmma_f32_16x16x32_f16 (512 halves/frag; lane: N=lane%16, K=8*(lane/16)+j, +8 for j>=8).
// ---------------------------------------------------------------------------
__global__ void pack_weights_kernel(const float* __restrict__ W1,
                                    const float* __restrict__ W2,
                                    _Float16* __restrict__ pw) {
  for (int i = blockIdx.x * blockDim.x + threadIdx.x; i < TOTAL_PACK;
       i += gridDim.x * blockDim.x) {
    if (i < OFF_W2) {
      int region = i / 5120;          // 0:q-fold 1:k-fold 2:qk
      int idx    = i % 5120;
      int j      = idx & 15;
      int lane   = (idx >> 4) & 31;
      int frag   = idx >> 9;          // 0..9
      int nt     = frag % 5;
      int s      = frag / 5;
      int n      = nt * 16 + (lane & 15);
      int kl     = s * 32 + j + ((lane < 16) ? 0 : 8) + ((j < 8) ? 0 : 8);   // 0..63
      float val;
      if (region == 0)      val = W1[kl * H1DIM + n] + W1[(128 + kl) * H1DIM + n];
      else if (region == 1) val = W1[(64 + kl) * H1DIM + n] - W1[(128 + kl) * H1DIM + n];
      else                  val = W1[(192 + kl) * H1DIM + n];
      pw[i] = (_Float16)val;
    } else {
      int idx  = i - OFF_W2;
      int j    = idx & 15;
      int lane = (idx >> 4) & 31;
      int frag = idx >> 9;            // 0..8
      int nt   = frag % 3;
      int s    = frag / 3;
      int n    = nt * 16 + (lane & 15);
      int kl   = s * 32 + j + ((lane < 16) ? 0 : 8) + ((j < 8) ? 0 : 8);     // 0..95
      float val = (kl < H1DIM && n < H2DIM) ? W2[kl * H2DIM + n] : 0.0f;
      pw[i] = (_Float16)val;
    }
  }
}

// ---------------------------------------------------------------------------
__device__ __forceinline__ v16h pack16(float4 a, float4 b, float4 c, float4 d) {
  v16h r;
  r[0]  = (_Float16)a.x; r[1]  = (_Float16)a.y; r[2]  = (_Float16)a.z; r[3]  = (_Float16)a.w;
  r[4]  = (_Float16)b.x; r[5]  = (_Float16)b.y; r[6]  = (_Float16)b.z; r[7]  = (_Float16)b.w;
  r[8]  = (_Float16)c.x; r[9]  = (_Float16)c.y; r[10] = (_Float16)c.z; r[11] = (_Float16)c.w;
  r[12] = (_Float16)d.x; r[13] = (_Float16)d.y; r[14] = (_Float16)d.z; r[15] = (_Float16)d.w;
  return r;
}
__device__ __forceinline__ float4 mul4(float4 a, float4 b) {
  return make_float4(a.x * b.x, a.y * b.y, a.z * b.z, a.w * b.w);
}

// Stage one 16x64 f32 k-tile into an LDS buffer (rows >= L clamped to L-1:
// garbage rows are harmless, GEMM rows are independent and their logits unused).
__device__ __forceinline__ void stage_tile(const float4* __restrict__ k4,
                                           float (*dst)[68],
                                           size_t kbase16, int l0, int L, int lane) {
#pragma unroll
  for (int i = 0; i < 8; ++i) {
    int flat = i * 32 + lane;
    int row  = flat >> 4;
    int c4   = flat & 15;
    int lg   = l0 + row;
    if (lg >= L) lg = L - 1;
    const float4* src = k4 + kbase16 + (size_t)lg * 16 + c4;
    float* d = &dst[row][c4 * 4];
#if HAVE_ASYNC
    async_cp16(src, d);
#else
    *(float4*)d = *src;
#endif
  }
}

// One folded-GEMM1 K-slab: acc = cin + k@(W1b-W1c) + (q*k)@W1d  (for this slab)
__device__ __forceinline__ void gemm1_slab(int s, const float* krow, const float* qs,
                                           const _Float16* __restrict__ pw,
                                           int lane, int halfsel,
                                           const v8f* cin, v8f* acc) {
  const int base = s * 32 + halfsel * 8;
  float4 k0 = *(const float4*)(krow + base);
  float4 k1 = *(const float4*)(krow + base + 4);
  float4 k2 = *(const float4*)(krow + base + 16);
  float4 k3 = *(const float4*)(krow + base + 20);
  v16h ak = pack16(k0, k1, k2, k3);
#pragma unroll
  for (int nt = 0; nt < 5; ++nt) {
    v16h bfr = *(const v16h*)(pw + OFF_K + (size_t)(s * 5 + nt) * 512 + lane * 16);
    acc[nt] = __builtin_amdgcn_wmma_f32_16x16x32_f16(
        false, ak, false, bfr, (short)0, cin[nt], false, false);
  }
  float4 q0 = *(const float4*)(qs + base);
  float4 q1 = *(const float4*)(qs + base + 4);
  float4 q2 = *(const float4*)(qs + base + 16);
  float4 q3 = *(const float4*)(qs + base + 20);
  v16h aqk = pack16(mul4(k0, q0), mul4(k1, q1), mul4(k2, q2), mul4(k3, q3));
#pragma unroll
  for (int nt = 0; nt < 5; ++nt) {
    v16h bfr = *(const v16h*)(pw + OFF_QK + (size_t)(s * 5 + nt) * 512 + lane * 16);
    acc[nt] = __builtin_amdgcn_wmma_f32_16x16x32_f16(
        false, aqk, false, bfr, (short)0, acc[nt], false, false);
  }
}

// ---------------------------------------------------------------------------
// Main kernel: 1 block per batch element, 128 threads (4 waves of 32).
// ---------------------------------------------------------------------------
__global__ __launch_bounds__(128)
void attn_main_kernel(const float* __restrict__ q,
                      const float4* __restrict__ k4,
                      const float* __restrict__ v,
                      const uint8_t* __restrict__ mask,
                      const float* __restrict__ b1,
                      const float* __restrict__ b2,
                      const float* __restrict__ wf,
                      const float* __restrict__ bfp,
                      const _Float16* __restrict__ pw,
                      float* __restrict__ out,
                      int L) {
  const int t    = threadIdx.x;
  const int wave = t >> 5;
  const int lane = t & 31;
  const int b    = blockIdx.x;

  __shared__ __align__(16) float q_s[EDIM];
  __shared__ float b1_s[H1DIM];
  __shared__ float b2_s[48];
  __shared__ float wf_s[H2DIM];
  __shared__ float bf_s;
  __shared__ __align__(16) float kt_s[NWAVE][2][16][68];  // double-buffered k tiles
  __shared__ __align__(16) float h1_s[NWAVE][16][100];    // cols 80..99 zero pad
  __shared__ __align__(16) float h2_s[NWAVE][16][52];
  __shared__ float logit_s[LPAD];
  __shared__ float red_s[128];
  __shared__ float osum_s[2][EDIM];

  // --- block-wide staging ---
  if (t < EDIM)  q_s[t]  = q[(size_t)b * EDIM + t];
  if (t < H1DIM) b1_s[t] = b1[t];
  if (t < 48)    b2_s[t] = (t < H2DIM) ? b2[t] : 0.0f;
  if (t < H2DIM) wf_s[t] = wf[t];
  if (t == 0)    bf_s = bfp[0];
  for (int i = t; i < LPAD; i += 128) logit_s[i] = -__builtin_inff();
  for (int i = t; i < NWAVE * 16 * 20; i += 128) {
    int col = 80 + (i % 20);
    int rw  = i / 20;
    h1_s[rw >> 4][rw & 15][col] = 0.0f;
  }
  __syncthreads();

  const int ntiles  = (L + 15) >> 4;
  const int m       = lane & 15;
  const int halfsel = (lane < 16) ? 0 : 1;
  const size_t kb16 = (size_t)b * L * 16;

  // ---- per-wave constant term: accq = b1 + q @ (W1a+W1c) ----
  v8f accq[5];
#pragma unroll
  for (int nt = 0; nt < 5; ++nt) {
    float bb = b1_s[nt * 16 + m];
#pragma unroll
    for (int r = 0; r < 8; ++r) accq[nt][r] = bb;
  }
#pragma unroll
  for (int s = 0; s < 2; ++s) {
    const int base = s * 32 + halfsel * 8;
    float4 q0 = *(const float4*)(q_s + base);
    float4 q1 = *(const float4*)(q_s + base + 4);
    float4 q2 = *(const float4*)(q_s + base + 16);
    float4 q3 = *(const float4*)(q_s + base + 20);
    v16h aq = pack16(q0, q1, q2, q3);
#pragma unroll
    for (int nt = 0; nt < 5; ++nt) {
      v16h bfr = *(const v16h*)(pw + OFF_Q + (size_t)(s * 5 + nt) * 512 + lane * 16);
      accq[nt] = __builtin_amdgcn_wmma_f32_16x16x32_f16(
          false, aq, false, bfr, (short)0, accq[nt], false, false);
    }
  }

  // ---- prologue: stage first tile into buffer 0 ----
  if (wave < ntiles)
    stage_tile(k4, kt_s[wave][0], kb16, wave << 4, L, lane);

  int cur = 0;
  for (int tile = wave; tile < ntiles; tile += NWAVE, cur ^= 1) {
    // prefetch next tile (or harmless dummy re-fetch) into the other buffer
    const int nxt = tile + NWAVE;
    const int l0n = ((nxt < ntiles) ? nxt : tile) << 4;
    stage_tile(k4, kt_s[wave][cur ^ 1], kb16, l0n, L, lane);
    // wait for the CURRENT tile's 8 transfers (in-order completion)
    TILE_SYNC(8);

    const int l0 = tile << 4;
    const float* krow = &kt_s[wave][cur][m][0];

    // ---- GEMM1 (folded, accq consumed as C of the first slab) ----
    v8f acc[5];
    gemm1_slab(0, krow, q_s, pw, lane, halfsel, accq, acc);
    gemm1_slab(1, krow, q_s, pw, lane, halfsel, acc,  acc);

    // ---- h1 -> LDS (C/D layout: row = r + 8*half, col = nt*16+m) ----
#pragma unroll
    for (int nt = 0; nt < 5; ++nt) {
#pragma unroll
      for (int r = 0; r < 8; ++r) {
        int row = r + halfsel * 8;
        h1_s[wave][row][nt * 16 + m] = acc[nt][r];
      }
    }
    CFENCE();

    // ---- GEMM2: h1[16,96pad] @ W2[96pad,48pad] ----
    v8f acc2[3] = {};
    const float* hrow = &h1_s[wave][m][0];
#pragma unroll
    for (int s = 0; s < 3; ++s) {
      const int base = s * 32 + halfsel * 8;
      float4 h0  = *(const float4*)(hrow + base);
      float4 h1v = *(const float4*)(hrow + base + 4);
      float4 h2v = *(const float4*)(hrow + base + 16);
      float4 h3  = *(const float4*)(hrow + base + 20);
      v16h a2 = pack16(h0, h1v, h2v, h3);
#pragma unroll
      for (int nt = 0; nt < 3; ++nt) {
        v16h bfr = *(const v16h*)(pw + OFF_W2 + (size_t)(s * 3 + nt) * 512 + lane * 16);
        acc2[nt] = __builtin_amdgcn_wmma_f32_16x16x32_f16(
            false, a2, false, bfr, (short)0, acc2[nt], false, false);
      }
    }

    // ---- sigmoid(h2 + b2) -> LDS ----
#pragma unroll
    for (int nt = 0; nt < 3; ++nt) {
#pragma unroll
      for (int r = 0; r < 8; ++r) {
        int row = r + halfsel * 8;
        int col = nt * 16 + m;
        float x = acc2[nt][r] + b2_s[col];
        h2_s[wave][row][col] = 1.0f / (1.0f + __expf(-x));
      }
    }
    CFENCE();

    // ---- logits: h2[16,40] . Wf + bf, apply mask ----
    if (lane < 16) {
      int lg = l0 + lane;
      if (lg < L) {
        float sum = bf_s;
#pragma unroll
        for (int j = 0; j < H2DIM; ++j) sum += h2_s[wave][lane][j] * wf_s[j];
        logit_s[lg] = mask[(size_t)b * L + lg] ? -__builtin_inff() : sum;
      }
    }
    CFENCE();
  }

  __syncthreads();

  // ---- softmax over L (padded entries are -inf -> exp 0) ----
  float lm = -__builtin_inff();
  for (int l = t; l < LPAD; l += 128) lm = fmaxf(lm, logit_s[l]);
  red_s[t] = lm;
  __syncthreads();
  for (int s = 64; s > 0; s >>= 1) {
    if (t < s) red_s[t] = fmaxf(red_s[t], red_s[t + s]);
    __syncthreads();
  }
  const float mx = red_s[0];
  __syncthreads();
  float lsum = 0.0f;
  for (int l = t; l < LPAD; l += 128) {
    float sc = __expf(logit_s[l] - mx);
    logit_s[l] = sc;
    lsum += sc;
  }
  red_s[t] = lsum;
  __syncthreads();
  for (int s = 64; s > 0; s >>= 1) {
    if (t < s) red_s[t] += red_s[t + s];
    __syncthreads();
  }
  const float inv = 1.0f / red_s[0];

  // ---- out[b,e] = (1/sum) * sum_l score_l * v[b,l,e]  (coalesced over e) ----
  const int e     = t & 63;
  const int hh    = t >> 6;
  const int halfL = (L + 1) >> 1;
  const int lbeg  = hh * halfL;
  const int lend  = min(lbeg + halfL, L);
  float accv = 0.0f;
  for (int l = lbeg; l < lend; ++l)
    accv += logit_s[l] * v[((size_t)b * L + l) * EDIM + e];
  osum_s[hh][e] = accv;
  __syncthreads();
  if (t < EDIM)
    out[(size_t)b * EDIM + t] = (osum_s[0][t] + osum_s[1][t]) * inv;
}

// ---------------------------------------------------------------------------
extern "C" void kernel_launch(void* const* d_in, const int* in_sizes, int n_in,
                              void* d_out, int out_size, void* d_ws, size_t ws_size,
                              hipStream_t stream) {
  const float*   q    = (const float*)d_in[0];
  const float*   k    = (const float*)d_in[1];
  const float*   v    = (const float*)d_in[2];
  const uint8_t* mask = (const uint8_t*)d_in[3];
  const float*   W1   = (const float*)d_in[4];
  const float*   b1   = (const float*)d_in[5];
  const float*   W2   = (const float*)d_in[6];
  const float*   b2   = (const float*)d_in[7];
  const float*   wf   = (const float*)d_in[8];
  const float*   bf   = (const float*)d_in[9];
  float*         out  = (float*)d_out;

  const int B = in_sizes[0] / EDIM;              // q is [B,1,E]
  const int L = in_sizes[1] / (B * EDIM);        // k is [B,L,E]

  _Float16* pw = (_Float16*)d_ws;                // ~40 KB packed weights

  pack_weights_kernel<<<32, 256, 0, stream>>>(W1, W2, pw);
  attn_main_kernel<<<B, 128, 0, stream>>>(q, (const float4*)k, v, mask,
                                          b1, b2, wf, bf, pw, out, L);
}
